// AnaphoricityScorer_63737314673218
// MI455X (gfx1250) — matrix operand order repaired and checked by reading, not compiled, not run on hardware
//
#include <hip/hip_runtime.h>
#include <stdint.h>

// ---------------------------------------------------------------------------
// Shapes from the reference
// ---------------------------------------------------------------------------
#define NM  8192          // N_MENTIONS
#define EMB 1024

typedef __bf16 bf16;
typedef __attribute__((ext_vector_type(16))) __bf16 v16bf;
typedef __attribute__((ext_vector_type(8)))  __bf16 v8bf;
typedef __attribute__((ext_vector_type(8)))  float  v8f;
typedef __attribute__((ext_vector_type(4)))  float  v4f;

// LDS tile geometry for the attention kernel: WG tile 256(M) x 128(N), K-step 32.
// Row stride 40 bf16 = 80B (64B data + 16B pad): row->bank phase = 20*r mod 64,
// distinct for r = 0..15  => conflict-free ds_load_b128.
#define LDS_STRIDE  40
#define A_ROWS      256
#define B_ROWS      128
#define A_BUF_ELems (A_ROWS * LDS_STRIDE)          // 10240 bf16 = 20 KiB
#define B_BUF_ELems (B_ROWS * LDS_STRIDE)          // 5120  bf16 = 10 KiB
#define KSTEPS      (EMB / 32)                      // 32

// ---------------------------------------------------------------------------
// WMMA fragment layouts (wave32), per CDNA5 ISA:
// A 16x32 (16-bit): lane L: row M=L%16; g=L>>4: elems 0..7 = K g*8..g*8+7,
//                   elems 8..15 = K 16+g*8..23+g*8.
// B 32x16 (16-bit): lane L: col N=L%16; elems 0..7 = K g*8..g*8+7,
//                   elems 8..15 = K 16+g*8..23+g*8.
// C/D 16x16 f32:    elem v: M = v + 8*g ; N = lane%16.
// ---------------------------------------------------------------------------
__device__ __forceinline__ v16bf combine16(v8bf lo, v8bf hi) {
    return __builtin_shufflevector(lo, hi, 0, 1, 2, 3, 4, 5, 6, 7,
                                          8, 9, 10, 11, 12, 13, 14, 15);
}

__device__ __forceinline__ v16bf load_a_frag_f32(const float* __restrict__ row,
                                                 int k0, int g) {
    v8f lo = *(const v8f*)(row + k0 + g * 8);
    v8f hi = *(const v8f*)(row + k0 + 16 + g * 8);
    v16bf a;
#pragma unroll
    for (int i = 0; i < 8; ++i) { a[i] = (bf16)lo[i]; a[i + 8] = (bf16)hi[i]; }
    return a;
}

__device__ __forceinline__ v16bf load_b_frag_f32(const float* __restrict__ row,
                                                 int k0, int g) {
    v8f lo = *(const v8f*)(row + k0 + g * 8);
    v8f hi = *(const v8f*)(row + k0 + 16 + g * 8);
    v16bf b;
#pragma unroll
    for (int i = 0; i < 8; ++i) { b[i] = (bf16)lo[i]; b[i + 8] = (bf16)hi[i]; }
    return b;
}

// ---------------------------------------------------------------------------
// Projection: out_bf16[m][n] = (sum_k X[m][k] * W[n][k] + b[n]) * scale
// grid = (EMB/128, NM/128, 2) ; z==0 -> Q (scale 1/32), z==1 -> K (scale 1)
// ---------------------------------------------------------------------------
__global__ __launch_bounds__(256) void proj_kernel(
    const float* __restrict__ X,
    const float* __restrict__ Wq, const float* __restrict__ bq,
    const float* __restrict__ Wk, const float* __restrict__ bk,
    bf16* __restrict__ Qout, bf16* __restrict__ Kout)
{
    const float* W; const float* bias; bf16* out; float scale;
    if (blockIdx.z == 0) { W = Wq; bias = bq; out = Qout; scale = 0.03125f; }
    else                 { W = Wk; bias = bk; out = Kout; scale = 1.0f;     }

    const int lane = threadIdx.x & 31;
    const int wave = threadIdx.x >> 5;
    const int r    = lane & 15;
    const int g    = lane >> 4;

    const int m0 = blockIdx.y * 128 + (wave >> 1) * 32;
    const int n0 = blockIdx.x * 128 + (wave & 1) * 64;

    v8f acc[2][4];
#pragma unroll
    for (int i = 0; i < 2; ++i)
#pragma unroll
        for (int j = 0; j < 4; ++j) acc[i][j] = (v8f)0.0f;

    const float* arow0 = X + (size_t)(m0 + r) * EMB;
    const float* arow1 = arow0 + (size_t)16 * EMB;

    for (int k0 = 0; k0 < EMB; k0 += 32) {
        v16bf a0 = load_a_frag_f32(arow0, k0, g);
        v16bf a1 = load_a_frag_f32(arow1, k0, g);
        v16bf bf[4];
#pragma unroll
        for (int j = 0; j < 4; ++j)
            bf[j] = load_b_frag_f32(W + (size_t)(n0 + j * 16 + r) * EMB, k0, g);
#pragma unroll
        for (int j = 0; j < 4; ++j) {
            acc[0][j] = __builtin_amdgcn_wmma_f32_16x16x32_bf16(
                false, a0, false, bf[j], (short)0, acc[0][j], false, false);
            acc[1][j] = __builtin_amdgcn_wmma_f32_16x16x32_bf16(
                false, a1, false, bf[j], (short)0, acc[1][j], false, false);
        }
    }

#pragma unroll
    for (int j = 0; j < 4; ++j) {
        const int n   = n0 + j * 16 + r;
        const float bv = bias[n];
#pragma unroll
        for (int i = 0; i < 2; ++i) {
#pragma unroll
            for (int v = 0; v < 8; ++v) {
                const int m = m0 + i * 16 + v + g * 8;
                out[(size_t)m * EMB + n] = (bf16)((acc[i][j][v] + bv) * scale);
            }
        }
    }
}

// ---------------------------------------------------------------------------
// Attention logits: out[m][n] = sum_k Q[m][k] * K[n][k] ; n > m -> -inf.
// grid = (NM/128, NM/256). WG tile 256x128; 8 waves of 64x64 (4x4 WMMA acc).
// Double-buffered async global->LDS pipeline (ASYNCcnt), WMMA from LDS.
// ---------------------------------------------------------------------------
__global__ __launch_bounds__(256) void attn_kernel(
    const bf16* __restrict__ Q, const bf16* __restrict__ Km,
    float* __restrict__ Out)
{
    const int wgN0 = blockIdx.x * 128;
    const int wgM0 = blockIdx.y * 256;
    const float NEGINF = -__builtin_inff();

    if (wgN0 > wgM0 + 255) {
        // Entire 256x128 block strictly above the diagonal: stream -inf.
        const int col = (threadIdx.x & 31) * 4;
        const int rb  = threadIdx.x >> 5;
        const v4f f = {NEGINF, NEGINF, NEGINF, NEGINF};
#pragma unroll
        for (int rph = 0; rph < 32; ++rph) {
            const int row = rb + rph * 8;
            __builtin_nontemporal_store(
                f, (v4f*)(Out + (size_t)(wgM0 + row) * NM + wgN0 + col));
        }
        return;
    }
    const bool needMask = (wgN0 + 127) > wgM0;

    __shared__ __align__(16) bf16 lds[2 * A_BUF_ELems + 2 * B_BUF_ELems]; // 60 KiB

    const int t    = threadIdx.x;
    const int lane = t & 31;
    const int wave = t >> 5;
    const int r    = lane & 15;
    const int g    = lane >> 4;

    const int mloc = (wave >> 1) * 64;   // 0/64/128/192
    const int nloc = (wave & 1) * 64;    // 0/64

    // ---- async copy of one K-stage (32 columns) into LDS buffer `buf` ----
    auto issue_stage = [&](int buf, int k0) {
        const int abase = buf * A_BUF_ELems;
        const int bbase = 2 * A_BUF_ELems + buf * B_BUF_ELems;
        // A tile: 256 rows * 4 chunks of 16B  -> 4 per thread
#pragma unroll
        for (int i = 0; i < 4; ++i) {
            const int idx = t + i * 256;
            const int row = idx >> 2, c = idx & 3;
            const bf16* gp = Q + (size_t)(wgM0 + row) * EMB + k0 + c * 8;
            const uint32_t lp =
                (uint32_t)(uintptr_t)&lds[abase + row * LDS_STRIDE + c * 8];
            asm volatile("global_load_async_to_lds_b128 %0, %1, off"
                         :: "v"(lp), "v"(gp) : "memory");
        }
        // B tile: 128 rows * 4 chunks of 16B  -> 2 per thread
#pragma unroll
        for (int i = 0; i < 2; ++i) {
            const int idx = t + i * 256;
            const int row = idx >> 2, c = idx & 3;
            const bf16* gp = Km + (size_t)(wgN0 + row) * EMB + k0 + c * 8;
            const uint32_t lp =
                (uint32_t)(uintptr_t)&lds[bbase + row * LDS_STRIDE + c * 8];
            asm volatile("global_load_async_to_lds_b128 %0, %1, off"
                         :: "v"(lp), "v"(gp) : "memory");
        }
    };

    v8f acc[4][4];
#pragma unroll
    for (int i = 0; i < 4; ++i)
#pragma unroll
        for (int j = 0; j < 4; ++j) acc[i][j] = (v8f)0.0f;

    issue_stage(0, 0);

    for (int ks = 0; ks < KSTEPS; ++ks) {
        if (ks + 1 < KSTEPS) {
            issue_stage((ks + 1) & 1, (ks + 1) * 32);
            // own 6 copies of stage ks done; 6 of stage ks+1 still in flight
            asm volatile("s_wait_asynccnt 6" ::: "memory");
        } else {
            asm volatile("s_wait_asynccnt 0" ::: "memory");
        }
        __syncthreads();   // all waves' stage-ks copies have landed in LDS

        const int abase = (ks & 1) * A_BUF_ELems;
        const int bbase = 2 * A_BUF_ELems + (ks & 1) * B_BUF_ELems;

        v16bf a[4], b[4];
#pragma unroll
        for (int ti = 0; ti < 4; ++ti) {
            const int row = mloc + ti * 16 + r;
            const int o   = abase + row * LDS_STRIDE;
            v8bf lo = *(const v8bf*)&lds[o + g * 8];
            v8bf hi = *(const v8bf*)&lds[o + 16 + g * 8];
            a[ti] = combine16(lo, hi);
        }
#pragma unroll
        for (int tj = 0; tj < 4; ++tj) {
            const int row = nloc + tj * 16 + r;
            const int o   = bbase + row * LDS_STRIDE;
            v8bf lo = *(const v8bf*)&lds[o + g * 8];
            v8bf hi = *(const v8bf*)&lds[o + 16 + g * 8];
            b[tj] = combine16(lo, hi);
        }
#pragma unroll
        for (int ti = 0; ti < 4; ++ti)
#pragma unroll
            for (int tj = 0; tj < 4; ++tj)
                acc[ti][tj] = __builtin_amdgcn_wmma_f32_16x16x32_bf16(
                    false, a[ti], false, b[tj], (short)0, acc[ti][tj],
                    false, false);

        __syncthreads();   // everyone finished reading buf[ks&1] before reuse
    }

    // ---- epilogue: causal mask + nontemporal f32 stores ----
#pragma unroll
    for (int ti = 0; ti < 4; ++ti) {
#pragma unroll
        for (int tj = 0; tj < 4; ++tj) {
            const int n = wgN0 + nloc + tj * 16 + r;
#pragma unroll
            for (int v = 0; v < 8; ++v) {
                const int m = wgM0 + mloc + ti * 16 + v + g * 8;
                float val = acc[ti][tj][v];
                if (needMask && (n > m)) val = NEGINF;
                __builtin_nontemporal_store(val, Out + (size_t)m * NM + n);
            }
        }
    }
}

// ---------------------------------------------------------------------------
// Host-side launcher.
// ---------------------------------------------------------------------------
extern "C" void kernel_launch(void* const* d_in, const int* in_sizes, int n_in,
                              void* d_out, int out_size, void* d_ws, size_t ws_size,
                              hipStream_t stream) {
    (void)in_sizes; (void)n_in; (void)out_size; (void)ws_size;
    const float* X  = (const float*)d_in[0];
    const float* Wq = (const float*)d_in[1];
    const float* bq = (const float*)d_in[2];
    const float* Wk = (const float*)d_in[3];
    const float* bk = (const float*)d_in[4];

    bf16* Qb = (bf16*)d_ws;
    bf16* Kb = Qb + (size_t)NM * EMB;           // 16 MiB each in d_ws

    dim3 pGrid(EMB / 128, NM / 128, 2);         // (8, 64, 2)
    proj_kernel<<<pGrid, 256, 0, stream>>>(X, Wq, bq, Wk, bk, Qb, Kb);

    dim3 aGrid(NM / 128, NM / 256);             // (64, 32)
    attn_kernel<<<aGrid, 256, 0, stream>>>(Qb, Kb, (float*)d_out);
}
